// RPN_53549652247337
// MI455X (gfx1250) — matrix operand order repaired and checked by reading, not compiled, not run on hardware
//
#include <hip/hip_runtime.h>

// ---------------- Types for WMMA ----------------
typedef __attribute__((ext_vector_type(16))) __bf16 v16bf;
typedef __attribute__((ext_vector_type(8)))  float  v8f;

union Frag { uint4 q[2]; v16bf v; };

__device__ __forceinline__ unsigned short f2bf(float f) {
  unsigned u = __float_as_uint(f);
  u += 0x7FFFu + ((u >> 16) & 1u);           // round-to-nearest-even
  return (unsigned short)(u >> 16);
}

// CDNA5 async global->LDS staging (ASYNCcnt path).
// LDS dest byte offset = low 32 bits of the generic pointer to shared memory
__device__ __forceinline__ void async_lds_b128(unsigned short* lds_dst,
                                               const unsigned short* gsrc) {
  unsigned d = (unsigned)(unsigned long long)lds_dst;
  unsigned long long a = (unsigned long long)gsrc;
  asm volatile("global_load_async_to_lds_b128 %0, %1, off"
               :: "v"(d), "v"(a) : "memory");
}
__device__ __forceinline__ void wait_async0() {
  asm volatile("s_wait_asynccnt 0x0" ::: "memory");
}

// ---------------- Problem constants ----------------
#define CIN   512
#define COUT  512
#define HH    128
#define WW    128
#define NPIX  (HH*WW)           // 16384
#define KTOT  (9*CIN)           // 4608
#define PADH  130
#define PADW  132
#define PADPLANE (PADH*PADW)    // 17160
#define NANCH 9
#define NBOX  (NPIX*NANCH)      // 147456
#define NPOOL 6000
#define NPOST 300

// ---------------- K0: pad + fp32->bf16 input ----------------
__global__ void k_pad(const float* __restrict__ in, unsigned short* __restrict__ out) {
  int idx = blockIdx.x * 256 + threadIdx.x;
  if (idx >= CIN * PADPLANE) return;
  int ci  = idx / PADPLANE;
  int rem = idx % PADPLANE;
  int yp = rem / PADW, xp = rem % PADW;
  unsigned short v = 0;
  if (yp >= 1 && yp <= HH && xp >= 1 && xp <= WW)
    v = f2bf(in[(size_t)ci * NPIX + (yp - 1) * WW + (xp - 1)]);
  out[idx] = v;
}

// ---------------- K1: reorder conv weights -> [co][kk*512+ci] bf16 ----------------
__global__ void k_cvt_wc(const float* __restrict__ w, unsigned short* __restrict__ out) {
  int idx = blockIdx.x * 256 + threadIdx.x;
  if (idx >= COUT * CIN * 9) return;
  int co  = idx / (CIN * 9);
  int rem = idx % (CIN * 9);
  int ci  = rem / 9;
  int kk  = rem % 9;                    // dy*3+dx
  out[(size_t)co * KTOT + kk * CIN + ci] = f2bf(w[idx]);
}

// ---------------- K2: pack loc+score weights -> [64][512] bf16 ----------------
__global__ void k_cvt_w2(const float* __restrict__ wloc, const float* __restrict__ wsc,
                         unsigned short* __restrict__ out) {
  int idx = blockIdx.x * 256 + threadIdx.x;
  if (idx >= 64 * 512) return;
  int m = idx / 512, c = idx % 512;
  float v = 0.0f;
  if (m < 36)       v = wloc[m * 512 + c];
  else if (m < 54)  v = wsc[(m - 36) * 512 + c];
  out[idx] = f2bf(v);
}

// ---------------- K3: 3x3 conv as implicit GEMM (bf16 WMMA) ----------------
// Grid: (4 M-tiles of 128 co, 128 image rows). Block: 256 thr = 8 waves.
__global__ __launch_bounds__(256) void k_conv_gemm(
    const unsigned short* __restrict__ inpad,   // [512][130][132] bf16
    const unsigned short* __restrict__ wc,      // [512][4608] bf16
    const float* __restrict__ bconv,
    unsigned short* __restrict__ hout)          // [512][16384] bf16
{
  __shared__ unsigned short As[128 * 32];
  __shared__ unsigned short Bs[128 * 32];
  const int tid  = threadIdx.x;
  const int lane = tid & 31;
  const int wv   = tid >> 5;
  const int wave_m = wv & 3;        // 4 waves along M (32 co each)
  const int wave_n = wv >> 2;       // 2 waves along N (64 px each)
  const int co0 = blockIdx.x * 128;
  const int y   = blockIdx.y;

  v8f acc[2][4] = {};

  const int ar = tid >> 1, ah = tid & 1;            // A staging role
  const int bc = tid & 31, bx0 = (tid >> 5) * 16;   // B staging role
  const unsigned short* arow = wc + (size_t)(co0 + ar) * KTOT + ah * 16;

  for (int kb = 0; kb < 144; ++kb) {                // K blocks of 32
    const int kk  = kb >> 4;                        // filter tap 0..8
    const int cib = kb & 15;                        // ci block 0..15
    const int dy = kk / 3, dx = kk % 3;

    // Stage A tile [128 co][32 k]: async global->LDS, no VGPR bounce
    async_lds_b128(As + ar * 32 + ah * 16,     arow + kb * 32);
    async_lds_b128(As + ar * 32 + ah * 16 + 8, arow + kb * 32 + 8);
    if (kb + 1 < 144)
      __builtin_prefetch(arow + (kb + 1) * 32, 0, 1);   // global_prefetch_b8

    // Stage B tile [128 x][32 ci] (im2col transpose of shifted input row)
    {
      const int ci = cib * 32 + bc;
      const unsigned short* src = inpad + (size_t)ci * PADPLANE + (y + dy) * PADW + dx;
      #pragma unroll
      for (int j = 0; j < 16; ++j) {
        int x = bx0 + j;
        Bs[x * 32 + bc] = src[x];
      }
    }
    wait_async0();          // A tile landed in LDS (per-wave ASYNCcnt)
    __syncthreads();        // (DScnt for B handled by compiler before barrier)

    const int hl = lane >> 4, cl = lane & 15;
    Frag a[2], b[4];
    #pragma unroll
    for (int am = 0; am < 2; ++am) {
      int row = wave_m * 32 + am * 16 + cl;
      a[am].q[0] = *(const uint4*)(As + row * 32 + hl * 16);
      a[am].q[1] = *(const uint4*)(As + row * 32 + hl * 16 + 8);
    }
    #pragma unroll
    for (int bn = 0; bn < 4; ++bn) {
      int n = wave_n * 64 + bn * 16 + cl;
      const uint4* p = (const uint4*)(Bs + n * 32 + hl * 16);
      b[bn].q[0] = p[0]; b[bn].q[1] = p[1];
    }
    #pragma unroll
    for (int am = 0; am < 2; ++am)
      #pragma unroll
      for (int bn = 0; bn < 4; ++bn)
        acc[am][bn] = __builtin_amdgcn_wmma_f32_16x16x32_bf16(
            false, a[am].v, false, b[bn].v, (short)0, acc[am][bn], false, false);
    __syncthreads();
  }

  // Epilogue: bias + ReLU, store bf16 h[co][p]
  const int hl = lane >> 4, cl = lane & 15;
  #pragma unroll
  for (int am = 0; am < 2; ++am)
    for (int bn = 0; bn < 4; ++bn)
      #pragma unroll
      for (int r = 0; r < 8; ++r) {
        int co = co0 + wave_m * 32 + am * 16 + r + 8 * hl;
        int x  = wave_n * 64 + bn * 16 + cl;
        float v = acc[am][bn][r] + bconv[co];
        v = fmaxf(v, 0.0f);
        hout[(size_t)co * NPIX + y * WW + x] = f2bf(v);
      }
}

// ---------------- K4: fused 1x1 loc+score GEMM (64 x 16384 x 512) ----------------
__global__ __launch_bounds__(256) void k_head_gemm(
    const unsigned short* __restrict__ hbf,    // [512][16384] bf16
    const unsigned short* __restrict__ w2,     // [64][512] bf16
    const float* __restrict__ bloc, const float* __restrict__ bsc,
    float* __restrict__ smallout)              // [64][16384] f32
{
  __shared__ unsigned short As[64 * 32];
  __shared__ unsigned short Bs[128 * 32];
  const int tid = threadIdx.x, lane = tid & 31, wv = tid >> 5;
  const int wave_m = wv & 3;       // 4 waves along M (16 each)
  const int wave_n = wv >> 2;      // 2 waves along N (64 each)
  const int n0 = blockIdx.x * 128;

  v8f acc[4] = {};

  for (int kt = 0; kt < 16; ++kt) {
    const int k0 = kt * 32;
    if (tid < 128) {               // A tile: async global->LDS
      int r = tid >> 1, h2 = tid & 1;
      const unsigned short* src = w2 + r * 512 + k0 + h2 * 16;
      async_lds_b128(As + r * 32 + h2 * 16,     src);
      async_lds_b128(As + r * 32 + h2 * 16 + 8, src + 8);
    }
    {
      int c = tid & 31, xg = (tid >> 5) * 16;
      const uint4* src = (const uint4*)(hbf + (size_t)(k0 + c) * NPIX + n0 + xg);
      union { uint4 q[2]; unsigned short s[16]; } tmp;
      tmp.q[0] = src[0]; tmp.q[1] = src[1];
      #pragma unroll
      for (int j = 0; j < 16; ++j) Bs[(xg + j) * 32 + c] = tmp.s[j];
    }
    wait_async0();
    __syncthreads();

    const int hl = lane >> 4, cl = lane & 15;
    Frag a, b;
    int row = wave_m * 16 + cl;
    a.q[0] = *(const uint4*)(As + row * 32 + hl * 16);
    a.q[1] = *(const uint4*)(As + row * 32 + hl * 16 + 8);
    #pragma unroll
    for (int bn = 0; bn < 4; ++bn) {
      const uint4* p = (const uint4*)(Bs + (wave_n * 64 + bn * 16 + cl) * 32 + hl * 16);
      b.q[0] = p[0]; b.q[1] = p[1];
      acc[bn] = __builtin_amdgcn_wmma_f32_16x16x32_bf16(
          false, a.v, false, b.v, (short)0, acc[bn], false, false);
    }
    __syncthreads();
  }

  const int hl = lane >> 4, cl = lane & 15;
  for (int bn = 0; bn < 4; ++bn)
    #pragma unroll
    for (int r = 0; r < 8; ++r) {
      int m = wave_m * 16 + r + 8 * hl;
      int n = n0 + wave_n * 64 + bn * 16 + cl;
      float bias = (m < 36) ? bloc[m] : ((m < 54) ? bsc[m - 36] : 0.0f);
      smallout[(size_t)m * NPIX + n] = acc[bn][r] + bias;
    }
}

// ---------------- K init: zero hist/ctl, pool scores = -inf ----------------
__global__ void k_init(int* hist, int* hist2, int* ctl, float* pool_s) {
  int i = blockIdx.x * 256 + threadIdx.x;
  if (i < 2048) { hist[i] = 0; hist2[i] = 0; }
  if (i < 8)    ctl[i] = 0;
  if (i < NPOOL) pool_s[i] = -__builtin_inff();
}

// ---------------- K5: softmax/decode/clip/filter + histogram ----------------
__global__ void k_decode(const float* __restrict__ smallout,
                         const int* __restrict__ ph, const int* __restrict__ pw,
                         float* __restrict__ scores, float* __restrict__ boxes,
                         int* __restrict__ hist) {
  int idx = blockIdx.x * 256 + threadIdx.x;
  if (idx >= NBOX) return;
  int p = idx / NANCH, a = idx % NANCH;
  int y = p >> 7, x = p & 127;
  float imgh = (float)ph[0], imgw = (float)pw[0];
  float stride = imgh / (float)HH;

  float dyv = smallout[(size_t)(a * 4 + 0) * NPIX + p];
  float dxv = smallout[(size_t)(a * 4 + 1) * NPIX + p];
  float dhv = smallout[(size_t)(a * 4 + 2) * NPIX + p];
  float dwv = smallout[(size_t)(a * 4 + 3) * NPIX + p];
  float s0  = smallout[(size_t)(36 + a * 2 + 0) * NPIX + p];
  float s1  = smallout[(size_t)(36 + a * 2 + 1) * NPIX + p];
  float fg = 1.0f / (1.0f + expf(s0 - s1));     // softmax fg prob

  float rr = (a < 3) ? 0.5f : ((a < 6) ? 1.0f : 2.0f);
  float ss = (float)(8 << (a % 3));
  float ha = 16.0f * ss * sqrtf(rr);
  float wa = 16.0f * ss * sqrtf(1.0f / rr);
  float cty = y * stride + 8.0f, ctx = x * stride + 8.0f;
  float cy = dyv * ha + cty, cx = dxv * wa + ctx;
  float bh = expf(dhv) * ha, bw = expf(dwv) * wa;
  float y1 = fminf(fmaxf(cy - 0.5f * bh, 0.0f), imgh);
  float x1 = fminf(fmaxf(cx - 0.5f * bw, 0.0f), imgw);
  float y2 = fminf(fmaxf(cy + 0.5f * bh, 0.0f), imgh);
  float x2 = fminf(fmaxf(cx + 0.5f * bw, 0.0f), imgw);
  boxes[idx * 4 + 0] = y1; boxes[idx * 4 + 1] = x1;
  boxes[idx * 4 + 2] = y2; boxes[idx * 4 + 3] = x2;

  bool keep = (y2 - y1 >= 16.0f) && (x2 - x1 >= 16.0f);
  float sc = keep ? fg : -__builtin_inff();
  scores[idx] = sc;
  if (keep) atomicAdd(&hist[__float_as_uint(fg) >> 21], 1);
}

// ---------------- K6a/b: two-level threshold scan for top-6000 ----------------
__global__ void k_thresh1(const int* __restrict__ hist, int* __restrict__ ctl) {
  if (threadIdx.x == 0 && blockIdx.x == 0) {
    int cum = 0, B = -1, above = 0;
    for (int b = 2047; b >= 0; --b) {
      int c = hist[b];
      if (cum + c >= NPOOL) { B = b; above = cum; break; }
      cum += c;
    }
    if (B < 0) above = cum;
    ctl[0] = B; ctl[2] = above;
  }
}
__global__ void k_hist2(const float* __restrict__ scores, const int* __restrict__ ctl,
                        int* __restrict__ hist2) {
  int idx = blockIdx.x * 256 + threadIdx.x;
  if (idx >= NBOX) return;
  float sc = scores[idx];
  if (!(sc >= 0.0f)) return;                      // -inf filtered
  unsigned u = __float_as_uint(sc);
  if ((int)(u >> 21) == ctl[0]) atomicAdd(&hist2[(u >> 10) & 0x7FF], 1);
}
__global__ void k_thresh2(const int* __restrict__ hist2, int* __restrict__ ctl) {
  if (threadIdx.x == 0 && blockIdx.x == 0) {
    int above1 = ctl[2];
    int cum = 0, B2 = -1;
    for (int b = 2047; b >= 0; --b) {
      int c = hist2[b];
      if (above1 + cum + c >= NPOOL) { B2 = b; break; }
      cum += c;
    }
    int ca = above1 + cum;
    ctl[1] = B2; ctl[2] = ca; ctl[3] = NPOOL - ca;
  }
}

// ---------------- K7: compact top-6000 pool ----------------
__global__ void k_compact(const float* __restrict__ scores, const float* __restrict__ boxes,
                          int* __restrict__ ctl,
                          float* __restrict__ pool_s, float* __restrict__ pool_b) {
  int idx = blockIdx.x * 256 + threadIdx.x;
  if (idx >= NBOX) return;
  float sc = scores[idx];
  if (!(sc >= 0.0f)) return;
  unsigned u = __float_as_uint(sc);
  int k1 = (int)(u >> 21), k2 = (int)((u >> 10) & 0x7FF);
  int B1 = ctl[0], B2 = ctl[1], ca = ctl[2], need = ctl[3];
  int pos = -1;
  if (k1 > B1 || (k1 == B1 && k2 > B2)) {
    pos = atomicAdd(&ctl[4], 1);
  } else if (k1 == B1 && k2 == B2) {
    int q = atomicAdd(&ctl[5], 1);
    if (q < need) pos = ca + q;
  }
  if (pos < 0 || pos >= NPOOL) return;
  pool_s[pos] = sc;
  pool_b[pos * 4 + 0] = boxes[idx * 4 + 0];
  pool_b[pos * 4 + 1] = boxes[idx * 4 + 1];
  pool_b[pos * 4 + 2] = boxes[idx * 4 + 2];
  pool_b[pos * 4 + 3] = boxes[idx * 4 + 3];
}

// ---------------- K8: single-block greedy NMS (300 rounds) ----------------
__global__ __launch_bounds__(1024) void k_nms(const float* __restrict__ pool_s,
                                              const float* __restrict__ pool_b,
                                              float* __restrict__ out) {
  __shared__ float ls[NPOOL];
  __shared__ float redv[1024];
  __shared__ int   redi[1024];
  __shared__ float selbox[4];
  __shared__ int   seljs;
  const int tid = threadIdx.x;
  const float NEG = -__builtin_inff();

  float by1[6], bx1[6], by2[6], bx2[6], bar[6];
  #pragma unroll
  for (int e = 0; e < 6; ++e) {
    int i = tid + e * 1024;
    if (i < NPOOL) {
      ls[i]  = pool_s[i];
      by1[e] = pool_b[i * 4 + 0]; bx1[e] = pool_b[i * 4 + 1];
      by2[e] = pool_b[i * 4 + 2]; bx2[e] = pool_b[i * 4 + 3];
      bar[e] = (by2[e] - by1[e]) * (bx2[e] - bx1[e]);
    }
  }
  __syncthreads();

  for (int it = 0; it < NPOST; ++it) {
    float m = NEG; int mi = 0x7fffffff;
    #pragma unroll
    for (int e = 0; e < 6; ++e) {
      int i = tid + e * 1024;
      if (i < NPOOL) {
        float s = ls[i];
        if (s > m || (s == m && i < mi)) { m = s; mi = i; }
      }
    }
    redv[tid] = m; redi[tid] = mi;
    __syncthreads();
    for (int off = 512; off > 0; off >>= 1) {
      if (tid < off) {
        float s2 = redv[tid + off]; int i2 = redi[tid + off];
        if (s2 > redv[tid] || (s2 == redv[tid] && i2 < redi[tid])) {
          redv[tid] = s2; redi[tid] = i2;
        }
      }
      __syncthreads();
    }
    if (tid == 0) {
      int j = redi[0]; float kv = redv[0];
      if (kv > NEG) {
        seljs = j;
        float a0 = pool_b[j * 4 + 0], a1 = pool_b[j * 4 + 1];
        float a2 = pool_b[j * 4 + 2], a3 = pool_b[j * 4 + 3];
        selbox[0] = a0; selbox[1] = a1; selbox[2] = a2; selbox[3] = a3;
        out[it * 4 + 0] = a0; out[it * 4 + 1] = a1;
        out[it * 4 + 2] = a2; out[it * 4 + 3] = a3;
      } else {
        seljs = -1;
        out[it * 4 + 0] = 0.0f; out[it * 4 + 1] = 0.0f;
        out[it * 4 + 2] = 0.0f; out[it * 4 + 3] = 0.0f;
      }
    }
    __syncthreads();
    if (seljs >= 0) {
      float sy1 = selbox[0], sx1 = selbox[1], sy2 = selbox[2], sx2 = selbox[3];
      float sarea = (sy2 - sy1) * (sx2 - sx1);
      #pragma unroll
      for (int e = 0; e < 6; ++e) {
        int i = tid + e * 1024;
        if (i < NPOOL) {
          float yy1 = fmaxf(sy1, by1[e]);
          float xx1 = fmaxf(sx1, bx1[e]);
          float yy2 = fminf(sy2, by2[e]);
          float xx2 = fminf(sx2, bx2[e]);
          float inter = fmaxf(yy2 - yy1, 0.0f) * fmaxf(xx2 - xx1, 0.0f);
          float iou = inter / (sarea + bar[e] - inter + 1e-9f);
          if (iou > 0.7f) ls[i] = NEG;
        }
      }
    }
    __syncthreads();
  }
}

// ---------------- Host launcher ----------------
extern "C" void kernel_launch(void* const* d_in, const int* in_sizes, int n_in,
                              void* d_out, int out_size, void* d_ws, size_t ws_size,
                              hipStream_t stream) {
  const float* fmap  = (const float*)d_in[0];
  const float* Wconv = (const float*)d_in[1];
  const float* bconv = (const float*)d_in[2];
  const float* Wloc  = (const float*)d_in[3];
  const float* bloc  = (const float*)d_in[4];
  const float* Wsc   = (const float*)d_in[5];
  const float* bsc   = (const float*)d_in[6];
  const int*   imh   = (const int*)d_in[7];
  const int*   imw   = (const int*)d_in[8];
  (void)in_sizes; (void)n_in; (void)out_size; (void)ws_size;

  char* ws = (char*)d_ws;
  size_t o = 0;
  auto take = [&](size_t bytes) { size_t r = o; o += (bytes + 255) & ~(size_t)255; return r; };
  unsigned short* inpad = (unsigned short*)(ws + take((size_t)CIN * PADPLANE * 2)); // 17.6 MB
  unsigned short* wcbf  = (unsigned short*)(ws + take((size_t)COUT * KTOT * 2));    //  4.7 MB
  unsigned short* hbf   = (unsigned short*)(ws + take((size_t)COUT * NPIX * 2));    // 16.8 MB
  unsigned short* w2bf  = (unsigned short*)(ws + take((size_t)64 * 512 * 2));
  float* smallout = (float*)(ws + take((size_t)64 * NPIX * 4));                     //  4.2 MB
  float* scores   = (float*)(ws + take((size_t)NBOX * 4));
  float* boxes    = (float*)(ws + take((size_t)NBOX * 4 * 4));
  int*   hist     = (int*)(ws + take(2048 * 4));
  int*   hist2    = (int*)(ws + take(2048 * 4));
  int*   ctl      = (int*)(ws + take(64));
  float* pool_s   = (float*)(ws + take((size_t)NPOOL * 4));
  float* pool_b   = (float*)(ws + take((size_t)NPOOL * 4 * 4));

  k_init<<<(NPOOL + 255) / 256, 256, 0, stream>>>(hist, hist2, ctl, pool_s);
  k_pad<<<(CIN * PADPLANE + 255) / 256, 256, 0, stream>>>(fmap, inpad);
  k_cvt_wc<<<(COUT * CIN * 9 + 255) / 256, 256, 0, stream>>>(Wconv, wcbf);
  k_cvt_w2<<<(64 * 512 + 255) / 256, 256, 0, stream>>>(Wloc, Wsc, w2bf);

  dim3 g3(COUT / 128, HH);
  k_conv_gemm<<<g3, 256, 0, stream>>>(inpad, wcbf, bconv, hbf);
  k_head_gemm<<<NPIX / 128, 256, 0, stream>>>(hbf, w2bf, bloc, bsc, smallout);

  k_decode<<<(NBOX + 255) / 256, 256, 0, stream>>>(smallout, imh, imw, scores, boxes, hist);
  k_thresh1<<<1, 1, 0, stream>>>(hist, ctl);
  k_hist2<<<(NBOX + 255) / 256, 256, 0, stream>>>(scores, ctl, hist2);
  k_thresh2<<<1, 1, 0, stream>>>(hist2, ctl);
  k_compact<<<(NBOX + 255) / 256, 256, 0, stream>>>(scores, boxes, ctl, pool_s, pool_b);
  k_nms<<<1, 1024, 0, stream>>>(pool_s, pool_b, (float*)d_out);
}
